// CombinedLoss_16638703305336
// MI455X (gfx1250) — compile-verified, matrix operand order checked by reading, not run on hardware
//
#include <hip/hip_runtime.h>
#include <math.h>

// ---------- types ----------
typedef __bf16 bf16;
typedef __attribute__((ext_vector_type(16))) __bf16 v16bf;
typedef __attribute__((ext_vector_type(8)))  float  v8f;

union Frag16 { v16bf v; bf16 e[16]; uint4 q[2]; };
union BF8    { uint4 q;  bf16 e[8]; };

#define NTOT  4096
#define DDIM  256
#define TILE  128
#define KSTEP 32
#define NT    32        // tiles per dimension (4096/128)
#define LDSB_STRIDE 40  // padded bf16 stride per staged B row (bank-conflict free)

__device__ __forceinline__ bf16 f2bf(float x) { return (bf16)x; }

// branchless (m, sum-exp) merge; never called with both m == -inf
__device__ __forceinline__ void mergeMS(float &m, float &s, float om, float os) {
    float nm = fmaxf(m, om);
    s = s * __expf(m - nm) + os * __expf(om - nm);
    m = nm;
}

// branchless online update with a single value (s starts at 0, m at -inf)
__device__ __forceinline__ void pushMS(float &m, float &s, float v) {
    float nm = fmaxf(m, v);
    s = s * __expf(m - nm) + __expf(v - nm);
    m = nm;
}

// fully branchless argmax merge (ties -> lower index, matching jnp.argmax);
// bitwise | and & avoid short-circuit control flow (no exec-mask juggling)
__device__ __forceinline__ void mergeArg(float &bv, int &bi, float ov, int oi) {
    const bool take = (ov > bv) | ((ov == bv) & (oi < bi));
    bv = take ? ov : bv;
    bi = take ? oi : bi;
}

// ---------- init scratch scalars ----------
__global__ void init_kernel(float* scal) {
    if (threadIdx.x < 8) scal[threadIdx.x] = 0.0f;
}

// ---------- fused GEMM + online reductions ----------
__global__ __launch_bounds__(256) void sim_gemm_kernel(
    const float* __restrict__ A, const float* __restrict__ B,
    float2* __restrict__ rowMS, float* __restrict__ rowArgV, int* __restrict__ rowArgI,
    float2* __restrict__ colMS, float* __restrict__ diagV, int* __restrict__ hardIdx)
{
    __shared__ __align__(16) bf16  ldsB[TILE * LDSB_STRIDE];  // 10 KB staged B tile (bf16)
    __shared__ __align__(16) float2 cbuf[8][8][16];           // column partial exchange

    const int tid     = threadIdx.x;
    const int wave    = tid >> 5;
    const int lane    = tid & 31;
    const int laneLow = lane & 15;
    const int laneHi  = lane >> 4;           // 0 or 1
    const int rowTile = blockIdx.y, colTile = blockIdx.x;
    const int rowBase = rowTile * TILE, colBase = colTile * TILE;

    v8f acc[8];
    {
        v8f z = {0.f,0.f,0.f,0.f,0.f,0.f,0.f,0.f};
        #pragma unroll
        for (int t = 0; t < 8; ++t) acc[t] = z;
    }

    // staging map: 2 threads per B row (col of sim), 16 K-values each
    const int cStage = tid >> 1;
    const int kHalf  = (tid & 1) * 16;
    const int rowA   = rowBase + wave * 16 + laneLow;   // A-operand row for this lane
    const int koff   = laneHi * 8;                      // K sub-offset per half-wave

    for (int kb = 0; kb < DDIM; kb += KSTEP) {
        // ---- stage B tile (f32 -> bf16) into LDS ----
        {
            const float4* src = (const float4*)(B + (size_t)(colBase + cStage) * DDIM + kb + kHalf);
            float4 f0 = src[0], f1 = src[1], f2 = src[2], f3 = src[3];
            BF8 p0, p1;
            p0.e[0]=f2bf(f0.x); p0.e[1]=f2bf(f0.y); p0.e[2]=f2bf(f0.z); p0.e[3]=f2bf(f0.w);
            p0.e[4]=f2bf(f1.x); p0.e[5]=f2bf(f1.y); p0.e[6]=f2bf(f1.z); p0.e[7]=f2bf(f1.w);
            p1.e[0]=f2bf(f2.x); p1.e[1]=f2bf(f2.y); p1.e[2]=f2bf(f2.z); p1.e[3]=f2bf(f2.w);
            p1.e[4]=f2bf(f3.x); p1.e[5]=f2bf(f3.y); p1.e[6]=f2bf(f3.z); p1.e[7]=f2bf(f3.w);
            uint4* dst = (uint4*)(ldsB + cStage * LDSB_STRIDE + kHalf);
            dst[0] = p0.q; dst[1] = p1.q;
        }
        __syncthreads();

        // ---- A fragment straight from global (L2-resident), convert to bf16 ----
        Frag16 af;
        {
            const float4* srcA = (const float4*)(A + (size_t)rowA * DDIM + kb + koff);
            float4 a0 = srcA[0], a1 = srcA[1];   // K = koff .. koff+7
            float4 a2 = srcA[4], a3 = srcA[5];   // K = koff+16 .. koff+23
            af.e[0]=f2bf(a0.x); af.e[1]=f2bf(a0.y); af.e[2]=f2bf(a0.z); af.e[3]=f2bf(a0.w);
            af.e[4]=f2bf(a1.x); af.e[5]=f2bf(a1.y); af.e[6]=f2bf(a1.z); af.e[7]=f2bf(a1.w);
            af.e[8]=f2bf(a2.x); af.e[9]=f2bf(a2.y); af.e[10]=f2bf(a2.z); af.e[11]=f2bf(a2.w);
            af.e[12]=f2bf(a3.x); af.e[13]=f2bf(a3.y); af.e[14]=f2bf(a3.z); af.e[15]=f2bf(a3.w);
        }

        // ---- 8 WMMA column tiles ----
        #pragma unroll
        for (int t = 0; t < 8; ++t) {
            Frag16 bfr;
            const bf16* bp = ldsB + (t * 16 + laneLow) * LDSB_STRIDE + koff;
            bfr.q[0] = *(const uint4*)(bp);
            bfr.q[1] = *(const uint4*)(bp + 16);
            acc[t] = __builtin_amdgcn_wmma_f32_16x16x32_bf16(
                false, af.v, false, bfr.v, (short)0, acc[t], false, false);
        }
        __syncthreads();
    }

    // ---- epilogue: branchless online softmax / argmax partials ----
    const float invT = 1.0f / 0.07f;
    const bool isDiag = (rowTile == colTile);

    float cm[8], cs[8];
    #pragma unroll
    for (int t = 0; t < 8; ++t) { cm[t] = -INFINITY; cs[t] = 0.f; }

    #pragma unroll
    for (int i = 0; i < 8; ++i) {
        const int row_g = rowBase + wave * 16 + laneHi * 8 + i;
        float m = -INFINITY, s = 0.f;
        float bv = -INFINITY; int bi = 0;       // row argmax (accuracy / pred)
        float hv = -INFINITY; int hi = 0;       // hard negative (excl. diagonal)
        float dv = -INFINITY;                   // diagonal value capture
        #pragma unroll
        for (int t = 0; t < 8; ++t) {
            const float v = acc[t][i] * invT;
            const int col_g = colBase + t * 16 + laneLow;
            const bool onDiag = (col_g == row_g);
            pushMS(m, s, v);
            mergeArg(bv, bi, v, col_g);
            mergeArg(hv, hi, onDiag ? -INFINITY : v, col_g);  // exclude diagonal
            dv = onDiag ? v : dv;                             // capture diag value
        }
        // butterfly across the 16-lane group sharing this row
        #pragma unroll
        for (int mk = 1; mk <= 8; mk <<= 1) {
            float om = __shfl_xor(m, mk, 32), os = __shfl_xor(s, mk, 32);
            mergeMS(m, s, om, os);
            float obv = __shfl_xor(bv, mk, 32); int obi = __shfl_xor(bi, mk, 32);
            mergeArg(bv, bi, obv, obi);
            float ohv = __shfl_xor(hv, mk, 32); int ohi = __shfl_xor(hi, mk, 32);
            mergeArg(hv, hi, ohv, ohi);
            dv = fmaxf(dv, __shfl_xor(dv, mk, 32));           // only one lane is finite
        }
        if (laneLow == 0) {
            const size_t rp = (size_t)row_g * NT + colTile;
            rowMS[rp]   = make_float2(m, s);
            rowArgV[rp] = bv;
            rowArgI[rp] = bi;
            if (isDiag) {
                hardIdx[row_g] = hi - colBase;
                diagV[row_g]   = dv;
            }
        }
    }

    // ---- column partials: scan rows per lane, merge halves, exchange via LDS ----
    #pragma unroll
    for (int i = 0; i < 8; ++i) {
        #pragma unroll
        for (int t = 0; t < 8; ++t) {
            pushMS(cm[t], cs[t], acc[t][i] * invT);
        }
    }
    #pragma unroll
    for (int t = 0; t < 8; ++t) {
        float om = __shfl_xor(cm[t], 16, 32), os = __shfl_xor(cs[t], 16, 32);
        mergeMS(cm[t], cs[t], om, os);
        if (laneHi == 0) cbuf[wave][t][laneLow] = make_float2(cm[t], cs[t]);
    }
    __syncthreads();
    if (tid < 128) {
        const int t = tid >> 4, c = tid & 15;
        float m = -INFINITY, s = 0.f;
        #pragma unroll
        for (int w = 0; w < 8; ++w) {
            float2 p = cbuf[w][t][c];
            mergeMS(m, s, p.x, p.y);
        }
        const int col_g = colBase + t * 16 + c;
        colMS[(size_t)col_g * NT + rowTile] = make_float2(m, s);
    }
}

// ---------- merge row partials: lse + pred ----------
__global__ void row_reduce_kernel(const float2* __restrict__ rowMS,
                                  const float* __restrict__ rowArgV,
                                  const int* __restrict__ rowArgI,
                                  const float* __restrict__ diagV,
                                  float* __restrict__ scal)
{
    const int r = blockIdx.x * blockDim.x + threadIdx.x;
    if (r >= NTOT) return;
    float m = -INFINITY, s = 0.f;
    float bv = -INFINITY; int bi = 0;
    #pragma unroll 4
    for (int ct = 0; ct < NT; ++ct) {
        float2 p = rowMS[(size_t)r * NT + ct];
        mergeMS(m, s, p.x, p.y);
        mergeArg(bv, bi, rowArgV[(size_t)r * NT + ct], rowArgI[(size_t)r * NT + ct]);
    }
    const float lse = m + __logf(s);
    atomicAdd(&scal[0], diagV[r] - lse);                    // sum of logp_A2B diag
    atomicAdd(&scal[2], (bi == r) ? 1.0f : 0.0f);           // accuracy count
}

// ---------- merge column partials: lse ----------
__global__ void col_reduce_kernel(const float2* __restrict__ colMS,
                                  const float* __restrict__ diagV,
                                  float* __restrict__ scal)
{
    const int c = blockIdx.x * blockDim.x + threadIdx.x;
    if (c >= NTOT) return;
    float m = -INFINITY, s = 0.f;
    #pragma unroll 4
    for (int rt = 0; rt < NT; ++rt) {
        float2 p = colMS[(size_t)c * NT + rt];
        mergeMS(m, s, p.x, p.y);
    }
    const float lse = m + __logf(s);
    atomicAdd(&scal[1], diagV[c] - lse);                    // sum of logp_B2A diag
}

// ---------- triplet loss: one wave per (b,k) pair ----------
__global__ __launch_bounds__(256) void triplet_kernel(
    const float* __restrict__ A, const float* __restrict__ B,
    const int* __restrict__ hardIdx, float* __restrict__ scal)
{
    const int wid  = (blockIdx.x * blockDim.x + threadIdx.x) >> 5;   // 0..4095
    const int lane = threadIdx.x & 31;
    const float* a = A + (size_t)wid * DDIM;
    const float* p = B + (size_t)wid * DDIM;
    const int  bidx = wid >> 7;
    const int  nrow = bidx * 128 + hardIdx[wid];
    const float* nn = B + (size_t)nrow * DDIM;

    float sp = 0.f, sn = 0.f;
    const int j0 = lane * 8;
    #pragma unroll
    for (int j = 0; j < 8; j += 4) {
        float4 av = *(const float4*)(a + j0 + j);
        float4 pv = *(const float4*)(p + j0 + j);
        float4 nv = *(const float4*)(nn + j0 + j);
        float d;
        d = av.x - pv.x + 1e-6f; sp += d * d;
        d = av.y - pv.y + 1e-6f; sp += d * d;
        d = av.z - pv.z + 1e-6f; sp += d * d;
        d = av.w - pv.w + 1e-6f; sp += d * d;
        d = av.x - nv.x + 1e-6f; sn += d * d;
        d = av.y - nv.y + 1e-6f; sn += d * d;
        d = av.z - nv.z + 1e-6f; sn += d * d;
        d = av.w - nv.w + 1e-6f; sn += d * d;
    }
    #pragma unroll
    for (int mk = 1; mk < 32; mk <<= 1) {
        sp += __shfl_xor(sp, mk, 32);
        sn += __shfl_xor(sn, mk, 32);
    }
    if (lane == 0) {
        const float l = sqrtf(sp) - sqrtf(sn) + 0.5f;
        atomicAdd(&scal[3], fmaxf(l, 0.0f));
    }
}

// ---------- finalize 4 scalar outputs ----------
__global__ void finalize_kernel(const float* __restrict__ scal, float* __restrict__ out)
{
    if (threadIdx.x == 0 && blockIdx.x == 0) {
        const float N = (float)NTOT;
        const float lossA = -scal[0] / N;
        const float lossB = -scal[1] / N;
        const float info  = 0.5f * (lossA + lossB);
        const float trip  = scal[3] / N;
        out[0] = info + 0.1f * trip;   // total_loss
        out[1] = info;                 // info_nce_loss
        out[2] = trip;                 // triplet_loss
        out[3] = scal[2] / N;          // accuracy
    }
}

extern "C" void kernel_launch(void* const* d_in, const int* in_sizes, int n_in,
                              void* d_out, int out_size, void* d_ws, size_t ws_size,
                              hipStream_t stream) {
    const float* A = (const float*)d_in[0];   // embeddings_A [32,128,256] f32
    const float* B = (const float*)d_in[1];   // embeddings_B [32,128,256] f32
    float* out = (float*)d_out;

    // workspace layout (~3 MB)
    char* w = (char*)d_ws;
    float2* rowMS   = (float2*)w; w += (size_t)NTOT * NT * sizeof(float2);
    float*  rowArgV = (float*) w; w += (size_t)NTOT * NT * sizeof(float);
    int*    rowArgI = (int*)   w; w += (size_t)NTOT * NT * sizeof(int);
    float2* colMS   = (float2*)w; w += (size_t)NTOT * NT * sizeof(float2);
    float*  diagV   = (float*) w; w += (size_t)NTOT * sizeof(float);
    int*    hardIdx = (int*)   w; w += (size_t)NTOT * sizeof(int);
    float*  scal    = (float*) w;

    init_kernel<<<1, 32, 0, stream>>>(scal);

    dim3 gGrid(NT, NT);
    sim_gemm_kernel<<<gGrid, 256, 0, stream>>>(A, B, rowMS, rowArgV, rowArgI,
                                               colMS, diagV, hardIdx);

    row_reduce_kernel<<<NTOT / 256, 256, 0, stream>>>(rowMS, rowArgV, rowArgI, diagV, scal);
    col_reduce_kernel<<<NTOT / 256, 256, 0, stream>>>(colMS, diagV, scal);

    triplet_kernel<<<(NTOT * 32) / 256, 256, 0, stream>>>(A, B, hardIdx, scal);

    finalize_kernel<<<1, 1, 0, stream>>>(scal, out);
}
